// StreamMultiheadAttention_33724083208865
// MI455X (gfx1250) — matrix operand order repaired and checked by reading, not compiled, not run on hardware
//
#include <hip/hip_runtime.h>
#include <hip/hip_bf16.h>

// ---------------- problem constants ----------------
#define BB   8
#define LQ   256
#define LV   576
#define PP   2048
#define LC   256
#define DM   1024
#define NH   16
#define HD   64
#define KT   2880          // LV + PP + LC
#define KOFF_PAST 576
#define KOFF_CUR  2624
#define ATT_SCALE 0.125f   // 1/sqrt(64)

typedef __attribute__((ext_vector_type(16))) __bf16 bf16x16;
typedef __attribute__((ext_vector_type(8)))  float  floatx8;

union Frag16 {
    bf16x16        v;
    unsigned short u[16];
    uint4          q[2];
};

__device__ __forceinline__ unsigned short f2bf(float f) {
    unsigned u = __builtin_bit_cast(unsigned, f);
    u = (u + 0x7FFFu + ((u >> 16) & 1u)) >> 16;    // round-to-nearest-even
    return (unsigned short)u;
}

__device__ __forceinline__ floatx8 wmma_bf16(bf16x16 a, bf16x16 b, floatx8 c) {
    return __builtin_amdgcn_wmma_f32_16x16x32_bf16(
        false, a, false, b, (short)0, c, false, false);
}

// =====================================================================
// Prepass 1: row-major f32 -> bf16 (activations), 2 elems/thread
// =====================================================================
__global__ __launch_bounds__(256) void cvt_rows_kernel(
    const float* __restrict__ src, unsigned short* __restrict__ dst)
{
    const size_t i = ((size_t)blockIdx.x * blockDim.x + threadIdx.x) * 2;
    const float2 v = *(const float2*)(src + i);
    *(unsigned*)(dst + i) = (unsigned)f2bf(v.x) | ((unsigned)f2bf(v.y) << 16);
}

// =====================================================================
// Prepass 2: W[k][n] f32 -> Wt[n][k] bf16 (LDS-tiled 32x32 transpose)
// =====================================================================
__global__ __launch_bounds__(256) void wtrans_kernel(
    const float* __restrict__ W, unsigned short* __restrict__ Wt)
{
    __shared__ float tile[32][33];
    const int n0 = blockIdx.x * 32, k0 = blockIdx.y * 32;
    const int tx = threadIdx.x, ty = threadIdx.y;
#pragma unroll
    for (int i = ty; i < 32; i += 8)
        tile[i][tx] = W[(size_t)(k0 + i) * DM + n0 + tx];
    __syncthreads();
#pragma unroll
    for (int i = ty; i < 32; i += 8)
        Wt[(size_t)(n0 + i) * DM + k0 + tx] = f2bf(tile[tx][i]);
}

// =====================================================================
// Prepass 3: fp32 past K/V [B,H,2048,64] -> bf16 at key offset 576
// =====================================================================
__global__ __launch_bounds__(256) void convert_past_kernel(
    const float* __restrict__ src, unsigned short* __restrict__ dst)
{
    const size_t idx = ((size_t)blockIdx.x * blockDim.x + threadIdx.x) * 2;
    const int d = (int)(idx & 63);
    size_t rest = idx >> 6;
    const int p = (int)(rest & (PP - 1)); rest >>= 11;
    const int h = (int)(rest & (NH - 1));
    const int b = (int)(rest >> 4);
    const float2 v = *(const float2*)(src + idx);
    const size_t di = (((size_t)b * NH + h) * KT + KOFF_PAST + p) * HD + d;
    *(unsigned*)(dst + di) = (unsigned)f2bf(v.x) | ((unsigned)f2bf(v.y) << 16);
}

// =====================================================================
// Pipelined GEMM core: one wave computes a 16x64 tile, double-buffered
// fragment loads so WMMA overlaps the next k-step's memory latency.
// =====================================================================
__device__ __forceinline__ void gemm_body(
    const unsigned short* __restrict__ arow,   // A row base (row-major bf16)
    const unsigned short* __restrict__ wtcol,  // Wt + ncol0*DM (col-major bf16)
    int lg, int ln, floatx8 acc[4])
{
    Frag16 a0, a1, b0[4], b1[4];
    auto loadAB = [&](int kb, Frag16& a, Frag16* bfr) {
        a.q[0] = *(const uint4*)(arow + kb + lg * 8);
        a.q[1] = *(const uint4*)(arow + kb + 16 + lg * 8);
#pragma unroll
        for (int t = 0; t < 4; ++t) {
            const unsigned short* wr = wtcol + (size_t)(t * 16 + ln) * DM + kb + lg * 8;
            bfr[t].q[0] = *(const uint4*)(wr);
            bfr[t].q[1] = *(const uint4*)(wr + 16);
        }
    };
    loadAB(0, a0, b0);
    for (int kb = 0; kb < DM; kb += 64) {
        loadAB(kb + 32, a1, b1);                       // next half-step
#pragma unroll
        for (int t = 0; t < 4; ++t) acc[t] = wmma_bf16(a0.v, b0[t].v, acc[t]);
        int kn = kb + 64; if (kn > DM - 32) kn = DM - 32;   // clamp (stay in ws)
        loadAB(kn, a0, b0);                            // step after next
#pragma unroll
        for (int t = 0; t < 4; ++t) acc[t] = wmma_bf16(a1.v, b1[t].v, acc[t]);
    }
}

// =====================================================================
// Projection GEMM: dst_bf16 = (Xbf @ Wt^T + bias) * scale, head-split out
// =====================================================================
__global__ __launch_bounds__(256) void proj_gemm_kernel(
    const unsigned short* __restrict__ Xbf, const unsigned short* __restrict__ Wt,
    const float* __restrict__ bias, unsigned short* __restrict__ dst,
    int rowsPerBatch, unsigned rowsMagic, int keyStride, int keyOffset, float scale)
{
    const int lane = threadIdx.x & 31;
    const int wid  = blockIdx.x * 8 + (threadIdx.x >> 5);
    const int mt   = wid >> 4;
    const int ng   = wid & 15;
    const int lg = lane >> 4, ln = lane & 15;

    const unsigned short* arow  = Xbf + (size_t)(mt * 16 + ln) * DM;
    const unsigned short* wtcol = Wt + (size_t)(ng * 64) * DM;

    floatx8 acc[4] = {};
    gemm_body(arow, wtcol, lg, ln, acc);

    // batch index via magic multiply (rows-per-batch | 16, tile stays in batch)
    const unsigned m0 = (unsigned)(mt * 16);
    const int b  = (int)(((unsigned long long)m0 * rowsMagic) >> 32);
    const int r0 = (int)m0 - b * rowsPerBatch;
#pragma unroll
    for (int t = 0; t < 4; ++t) {
        const int n = ng * 64 + t * 16 + ln;
        const int h = n >> 6, d = n & 63;
        const float bi = bias[n];
#pragma unroll
        for (int r = 0; r < 8; ++r) {
            const int key = keyOffset + r0 + r + 8 * lg;
            const size_t di = ((((size_t)b * NH + h) * keyStride) + key) * HD + d;
            dst[di] = f2bf((acc[t][r] + bi) * scale);
        }
    }
}

__device__ __forceinline__ bool key_valid(int key, int pvl,
                                          const unsigned char* __restrict__ cm) {
    if (key < KOFF_PAST) return true;
    if (key < KOFF_CUR)  return (key - KOFF_PAST) < pvl;
    return cm[key - KOFF_CUR] != 0;
}

// =====================================================================
// Fused flash attention, software-pipelined: each step issues the NEXT
// key block's K fragments + V rows before computing the current block.
// =====================================================================
__global__ __launch_bounds__(256) void attn_kernel(
    const unsigned short* __restrict__ Qbf, const unsigned short* __restrict__ Kbf,
    const unsigned short* __restrict__ Vbf, unsigned short* __restrict__ Obf,
    const int* __restrict__ past_valid_len, const unsigned char* __restrict__ cur_mask)
{
    __shared__ __align__(16) unsigned short sP[8][16 * 32];  // P tile, bf16
    __shared__ __align__(16) unsigned short sV[8][32 * 64];  // V tile, bf16

    const int lane = threadIdx.x & 31;
    const int w    = threadIdx.x >> 5;
    const int wid  = blockIdx.x * 8 + w;
    const int b  = wid >> 8;
    const int h  = (wid >> 4) & 15;
    const int qt = wid & 15;
    const int lg = lane >> 4, ln = lane & 15;
    const size_t bh = (size_t)b * NH + h;

    const unsigned short* qrow = Qbf + (bh * LQ + (qt * 16 + ln)) * HD;
    Frag16 qa0, qa1;
    qa0.q[0] = *(const uint4*)(qrow + lg * 16);
    qa0.q[1] = *(const uint4*)(qrow + lg * 16 + 8);
    qa1.q[0] = *(const uint4*)(qrow + 32 + lg * 16);
    qa1.q[1] = *(const uint4*)(qrow + 32 + lg * 16 + 8);

    const unsigned short* kbase = Kbf + bh * (size_t)KT * HD;
    const unsigned short* vbase = Vbf + bh * (size_t)KT * HD;
    const int pvl = past_valid_len[b];
    const unsigned char* cm = cur_mask + (size_t)b * LC;

    float mrow[8], lrow[8];
#pragma unroll
    for (int r = 0; r < 8; ++r) { mrow[r] = -__builtin_inff(); lrow[r] = 0.f; }
    floatx8 acc[4] = {};
    unsigned short* pbuf = sP[w];
    unsigned short* vb   = sV[w];

    // K fragments: [tile t][k-slice] ; double-buffered across steps
    Frag16 kA[4], kB[4];
    auto loadK = [&](int k0, Frag16* kf) {
#pragma unroll
        for (int t = 0; t < 2; ++t) {
            const unsigned short* kr = kbase + (size_t)(k0 + t * 16 + ln) * HD;
            kf[2 * t + 0].q[0] = *(const uint4*)(kr + lg * 16);
            kf[2 * t + 0].q[1] = *(const uint4*)(kr + lg * 16 + 8);
            kf[2 * t + 1].q[0] = *(const uint4*)(kr + 32 + lg * 16);
            kf[2 * t + 1].q[1] = *(const uint4*)(kr + 32 + lg * 16 + 8);
        }
    };
    auto loadV = [&](int k0, uint4* vr) {
        const uint4* src = (const uint4*)(vbase + (size_t)(k0 + lane) * HD);
#pragma unroll
        for (int i = 0; i < 8; ++i) vr[i] = src[i];
    };

    // one fused step: prefetch block kn, compute block k0 with kcur
    auto step = [&](int k0, Frag16* kcur, Frag16* knxt) {
        int kn = k0 + 32; if (kn > KT - 32) kn = KT - 32;
        loadK(kn, knxt);                       // next K fragments (global b128)
        uint4 vr[8];
        loadV(kn, vr);                         // next V rows into registers
        if (k0 + 64 < KT)                      // reach one more block into L2
            __builtin_prefetch(kbase + (size_t)(k0 + 64 + lane) * HD, 0, 3);

        // ---- scores for current block ----
        floatx8 s[2];
#pragma unroll
        for (int t = 0; t < 2; ++t) {
            floatx8 z = {};
            z = wmma_bf16(qa0.v, kcur[2 * t + 0].v, z);
            z = wmma_bf16(qa1.v, kcur[2 * t + 1].v, z);
            if (!key_valid(k0 + t * 16 + ln, pvl, cm)) {
#pragma unroll
                for (int r = 0; r < 8; ++r) z[r] = -__builtin_inff();
            }
            s[t] = z;
        }
        // ---- online softmax ----
#pragma unroll
        for (int r = 0; r < 8; ++r) {
            float bm = fmaxf(s[0][r], s[1][r]);
#pragma unroll
            for (int o = 8; o; o >>= 1) bm = fmaxf(bm, __shfl_xor(bm, o, 32));
            const float mn    = fmaxf(mrow[r], bm);
            const float alpha = __expf(mrow[r] - mn);
            mrow[r] = mn;
            const float p0 = __expf(s[0][r] - mn);
            const float p1 = __expf(s[1][r] - mn);
            s[0][r] = p0; s[1][r] = p1;
            float bs = p0 + p1;
#pragma unroll
            for (int o = 8; o; o >>= 1) bs += __shfl_xor(bs, o, 32);
            lrow[r] = lrow[r] * alpha + bs;
            acc[0][r] *= alpha; acc[1][r] *= alpha;
            acc[2][r] *= alpha; acc[3][r] *= alpha;
        }
        // ---- P (bf16) through LDS: C-layout -> A-fragment layout ----
#pragma unroll
        for (int t = 0; t < 2; ++t)
#pragma unroll
            for (int r = 0; r < 8; ++r)
                pbuf[(r + 8 * lg) * 32 + t * 16 + ln] = f2bf(s[t][r]);
        asm volatile("" ::: "memory");   // keep LDS issue order (in-order DS pipe)
        Frag16 pa;
        pa.q[0] = *(const uint4*)(pbuf + ln * 32 + lg * 16);
        pa.q[1] = *(const uint4*)(pbuf + ln * 32 + lg * 16 + 8);
        // ---- O += P @ V (reads current V tile from LDS) ----
#pragma unroll
        for (int t = 0; t < 4; ++t) {
            Frag16 vf;
            const int d = t * 16 + ln;
#pragma unroll
            for (int j = 0; j < 16; ++j) vf.u[j] = vb[(lg * 16 + j) * HD + d];
            acc[t] = wmma_bf16(pa.v, vf.v, acc[t]);
        }
        asm volatile("" ::: "memory");
        // ---- commit prefetched V rows to LDS for the next step ----
        {
            uint4* dv = (uint4*)(vb + lane * HD);
#pragma unroll
            for (int i = 0; i < 8; ++i) dv[i] = vr[i];
        }
    };

    // prologue: K block 0 + V block 0 staged
    loadK(0, kA);
    {
        uint4 vr[8];
        loadV(0, vr);
        uint4* dv = (uint4*)(vb + lane * HD);
#pragma unroll
        for (int i = 0; i < 8; ++i) dv[i] = vr[i];
    }
    for (int k0 = 0; k0 < KT; k0 += 64) {      // KT/32 = 90 blocks, even
        step(k0, kA, kB);
        step(k0 + 32, kB, kA);
    }
    // ---- normalize, write head-merged bf16 output [B,Lq,1024] ----
#pragma unroll
    for (int t = 0; t < 4; ++t) {
        const int d = t * 16 + ln;
#pragma unroll
        for (int r = 0; r < 8; ++r) {
            const int q = qt * 16 + r + 8 * lg;
            Obf[((size_t)b * LQ + q) * DM + h * HD + d] = f2bf(acc[t][r] / lrow[r]);
        }
    }
}

// =====================================================================
// Output GEMM: d_out = Obf[2048,1024](bf16) @ Wo(bf16,transposed) + bo
// =====================================================================
__global__ __launch_bounds__(256) void out_gemm_kernel(
    const unsigned short* __restrict__ Abf, const unsigned short* __restrict__ Wt,
    const float* __restrict__ bias, float* __restrict__ out)
{
    const int lane = threadIdx.x & 31;
    const int wid  = blockIdx.x * 8 + (threadIdx.x >> 5);
    const int mt   = wid >> 4;
    const int ng   = wid & 15;
    const int lg = lane >> 4, ln = lane & 15;

    const unsigned short* arow  = Abf + (size_t)(mt * 16 + ln) * DM;
    const unsigned short* wtcol = Wt + (size_t)(ng * 64) * DM;

    floatx8 acc[4] = {};
    gemm_body(arow, wtcol, lg, ln, acc);

#pragma unroll
    for (int t = 0; t < 4; ++t) {
        const int n = ng * 64 + t * 16 + ln;
        const float bi = bias[n];
#pragma unroll
        for (int r = 0; r < 8; ++r) {
            const int m = mt * 16 + r + 8 * lg;
            out[(size_t)m * DM + n] = acc[t][r] + bi;
        }
    }
}

// =====================================================================
extern "C" void kernel_launch(void* const* d_in, const int* in_sizes, int n_in,
                              void* d_out, int out_size, void* d_ws, size_t ws_size,
                              hipStream_t stream)
{
    const float* query   = (const float*)d_in[0];
    const float* visual  = (const float*)d_in[1];
    const float* pastK   = (const float*)d_in[2];
    const float* pastV   = (const float*)d_in[3];
    const int*   pvl     = (const int*)d_in[4];
    const float* curctx  = (const float*)d_in[5];
    const unsigned char* curmask = (const unsigned char*)d_in[6];  // numpy bool = 1B
    const float* Wq = (const float*)d_in[7];   const float* bq = (const float*)d_in[8];
    const float* Wk = (const float*)d_in[9];   const float* bk = (const float*)d_in[10];
    const float* Wv = (const float*)d_in[11];  const float* bv = (const float*)d_in[12];
    const float* Wo = (const float*)d_in[13];  const float* bo = (const float*)d_in[14];
    float* out = (float*)d_out;
    (void)in_sizes; (void)n_in; (void)out_size; (void)ws_size;

    // ---- workspace layout (bf16) ----
    const size_t QBF = (size_t)BB * NH * LQ * HD;   // 2,097,152
    const size_t KBF = (size_t)BB * NH * KT * HD;   // 23,592,960
    const size_t XQ  = (size_t)BB * LQ * DM;        // 2,097,152
    const size_t XV  = (size_t)BB * LV * DM;        // 4,718,592
    const size_t XC  = (size_t)BB * LC * DM;        // 2,097,152
    const size_t WT  = (size_t)DM * DM;             // 1,048,576
    unsigned short* qbf = (unsigned short*)d_ws;
    unsigned short* kbf = qbf + QBF;
    unsigned short* vbf = kbf + KBF;
    unsigned short* obf = vbf + KBF;
    unsigned short* xq  = obf + QBF;
    unsigned short* xv  = xq + XQ;
    unsigned short* xc  = xv + XV;
    unsigned short* wqT = xc + XC;
    unsigned short* wkT = wqT + WT;
    unsigned short* wvT = wkT + WT;
    unsigned short* woT = wvT + WT;

    // magic multipliers for batch index: b = (m * magic) >> 32
    const unsigned magic256 = (unsigned)(((1ULL << 32) + 255) / 256);
    const unsigned magic576 = (unsigned)(((1ULL << 32) + 575) / 576);

    // ---- prepass: convert activations + transpose/convert weights ----
    cvt_rows_kernel<<<(int)(XQ / 512), 256, 0, stream>>>(query,  xq);
    cvt_rows_kernel<<<(int)(XV / 512), 256, 0, stream>>>(visual, xv);
    cvt_rows_kernel<<<(int)(XC / 512), 256, 0, stream>>>(curctx, xc);
    dim3 tg(32, 32), tb(32, 8);
    wtrans_kernel<<<tg, tb, 0, stream>>>(Wq, wqT);
    wtrans_kernel<<<tg, tb, 0, stream>>>(Wk, wkT);
    wtrans_kernel<<<tg, tb, 0, stream>>>(Wv, wvT);
    wtrans_kernel<<<tg, tb, 0, stream>>>(Wo, woT);
    convert_past_kernel<<<32768, 256, 0, stream>>>(pastK, kbf);
    convert_past_kernel<<<32768, 256, 0, stream>>>(pastV, vbf);

    // ---- projections (waves = (M/16)*16, 8 waves / block, exact grids) ----
    proj_gemm_kernel<<<256, 256, 0, stream>>>(xq, wqT, bq, qbf, LQ, magic256, LQ, 0,        ATT_SCALE);
    proj_gemm_kernel<<<576, 256, 0, stream>>>(xv, wkT, bk, kbf, LV, magic576, KT, 0,        1.0f);
    proj_gemm_kernel<<<576, 256, 0, stream>>>(xv, wvT, bv, vbf, LV, magic576, KT, 0,        1.0f);
    proj_gemm_kernel<<<256, 256, 0, stream>>>(xc, wkT, bk, kbf, LC, magic256, KT, KOFF_CUR, 1.0f);
    proj_gemm_kernel<<<256, 256, 0, stream>>>(xc, wvT, bv, vbf, LC, magic256, KT, KOFF_CUR, 1.0f);

    // ---- fused flash attention: 2048 waves = 256 blocks ----
    attn_kernel<<<256, 256, 0, stream>>>(qbf, kbf, vbf, obf, pvl, curmask);

    // ---- output projection -> f32 d_out ----
    out_gemm_kernel<<<256, 256, 0, stream>>>(obf, woT, bo, out);
}
